// BASE_LIF_SNN_17575006175869
// MI455X (gfx1250) — compile-verified, matrix operand order checked by reading, not run on hardware
//
#include <hip/hip_runtime.h>

typedef __attribute__((ext_vector_type(2)))  float  v2f;
typedef __attribute__((ext_vector_type(8)))  float  v8f;
typedef __attribute__((ext_vector_type(8)))  __bf16 v8bf;
typedef __attribute__((ext_vector_type(16))) __bf16 v16bf;

#define T_STEPS 1000
#define BATCH   64
#define NN      512

static __device__ __forceinline__ v8f wmma_f32_k4(v2f a, v2f b, v8f c) {
  // exact fp32 matrix op: D = A(16x4) x B(4x16) + C
  return __builtin_amdgcn_wmma_f32_16x16x4_f32(false, a, false, b, (short)0, c,
                                               false, false);
}

static __device__ __forceinline__ v8f wmma_bf16_k32(v16bf a, v16bf b, v8f c) {
  // D(f32 16x16) = A(bf16 16x32) x B(bf16 32x16) + C(f32)
  return __builtin_amdgcn_wmma_f32_16x16x32_bf16(false, a, false, b, (short)0,
                                                 c, false, false);
}

static __device__ __forceinline__ v16bf cat8(v8bf a, v8bf b) {
  return __builtin_shufflevector(a, b, 0, 1, 2, 3, 4, 5, 6, 7, 8, 9, 10, 11,
                                 12, 13, 14, 15);
}

// ---------------------------------------------------------------------------
// Kernel 0: split W2 into bf16 hi + bf16 residual (into d_ws).  Since the
// spike matrix is exactly {0,1} in bf16, s@W2hi^T + s@W2lo^T with f32 WMMA
// accumulation reproduces the f32 GEMM to ~2^-17 relative error.
// ---------------------------------------------------------------------------
__global__ __launch_bounds__(256) void w2_split_kernel(
    const float* __restrict__ W2, __bf16* __restrict__ w2hi,
    __bf16* __restrict__ w2lo) {
  const int i = blockIdx.x * 256 + threadIdx.x;  // 512*512 elements
  const float w = W2[i];
  const __bf16 h = (__bf16)w;
  w2hi[i] = h;
  w2lo[i] = (__bf16)(w - (float)h);
}

// ---------------------------------------------------------------------------
// Kernel 1: ff[t,b,:] = x[b,t,:] @ W1^T — fully parallel, exact f32 WMMA
// (spike generation is a threshold crossing; keep this path bit-accurate).
// Staged into the states_snn[t,0] slot, overwritten in-place with v later.
// ---------------------------------------------------------------------------
__global__ __launch_bounds__(256) void ff_gemm_kernel(
    const float* __restrict__ X, const float* __restrict__ W1,
    float* __restrict__ states_out) {
  const int gwave  = (blockIdx.x * 256 + (int)threadIdx.x) >> 5;
  const int lane   = threadIdx.x & 31;
  const int tile_m = gwave >> 5;        // 4000 M-tiles
  const int tile_n = gwave & 31;        // 32 N-tiles
  const int m0 = tile_m << 4;
  const int n0 = tile_n << 4;
  const int lm   = lane & 15;
  const int hi   = lane >> 4;
  const int koff = hi << 1;

  const float* __restrict__ arow = X  + (size_t)(m0 + lm) * NN + koff;
  const float* __restrict__ brow = W1 + (size_t)(n0 + lm) * NN + koff;

  v8f acc = {};
#pragma unroll 8
  for (int k = 0; k < NN; k += 4) {
    v2f a = *(const v2f*)(arow + k);
    v2f b = *(const v2f*)(brow + k);
    acc = wmma_f32_k4(a, b, acc);
  }

  const int mbase = m0 + (hi << 3);
#pragma unroll
  for (int j = 0; j < 8; ++j) {
    const int m = mbase + j;
    const int b = m / T_STEPS;
    const int t = m - b * T_STEPS;
    states_out[((size_t)(t * 2) * BATCH + b) * NN + (n0 + lm)] = acc[j];
  }
}

// ---------------------------------------------------------------------------
// Kernel 2: sequential scan. 4 workgroups (16 batch rows each), 32 waves;
// wave w owns N-columns [16w,16w+16).  f32 elementwise spike dynamics; the
// recurrent readout GEMM u += s @ W2^T runs on v_wmma_f32_16x16x32_bf16 via
// the exact-bf16 spike panel (LDS ping-pong, 1 barrier/step) and the
// hi/lo-split weights (L2-resident, untransposed b128 loads).  The next
// step's ff lines are prefetched (global_prefetch_b8) behind the GEMM so the
// HBM latency of the staged ff stream is off the 1000-step critical path.
// ---------------------------------------------------------------------------
__global__ __launch_bounds__(1024) void snn_scan_kernel(
    const float* __restrict__ state_snn, const float* __restrict__ state_LI,
    const __bf16* __restrict__ w2hi, const __bf16* __restrict__ w2lo,
    const float* __restrict__ leak_v, const float* __restrict__ leak_i,
    const float* __restrict__ thresh, float* __restrict__ spikes_out,
    float* __restrict__ states_out, float* __restrict__ decoded_out) {
  __shared__ __bf16 s_lds[2][16 * NN];  // 2 x 16KB ping-pong spike panels

  const int wave  = threadIdx.x >> 5;   // 0..31 -> N tile
  const int lane  = threadIdx.x & 31;
  const int lm    = lane & 15;
  const int hi    = lane >> 4;
  const int btile = blockIdx.x << 4;
  const int n     = (wave << 4) + lm;

  const float lv = leak_v[n];
  const float li = leak_i[n];
  const float th = thresh[n];

  v8f v, s, u;
#pragma unroll
  for (int j = 0; j < 8; ++j) {
    const int m = btile + (hi << 3) + j;
    v[j] = state_snn[(size_t)m * NN + n];
    s[j] = state_snn[(size_t)(BATCH + m) * NN + n];
    u[j] = state_LI[(size_t)m * NN + n];
  }

  // bf16 B operand rows (column n of W2^T): contiguous in k -> b128 loads
  const __bf16* __restrict__ brow_hi = w2hi + (size_t)n * NN + (hi << 3);
  const __bf16* __restrict__ brow_lo = w2lo + (size_t)n * NN + (hi << 3);

  for (int t = 0; t < T_STEPS; ++t) {
    const size_t vbase = ((size_t)(t * 2) * BATCH) * NN;

    // exact f32 leak + reset-to-zero + spike; emit v, s, spikes
#pragma unroll
    for (int j = 0; j < 8; ++j) {
      const int m = btile + (hi << 3) + j;
      const size_t idx = vbase + (size_t)m * NN + n;
      const float ff = states_out[idx];              // staged by kernel 1
      v[j] = lv * v[j] * (1.0f - s[j]) + ff;
      s[j] = (v[j] - th) > 0.0f ? 1.0f : 0.0f;
      states_out[idx] = v[j];                        // states[t,0]
      states_out[idx + (size_t)BATCH * NN] = s[j];   // states[t,1]
      spikes_out[((size_t)t * BATCH + m) * NN + n] = s[j];
    }

    // publish spike tile as bf16 (exact for {0,1})
    __bf16* sl = s_lds[t & 1];
#pragma unroll
    for (int j = 0; j < 8; ++j)
      sl[((hi << 3) + j) * NN + n] = (__bf16)s[j];

    // prefetch next step's ff tile (hidden behind this step's GEMM)
    if (t + 1 < T_STEPS) {
      const size_t vnext = vbase + (size_t)2 * BATCH * NN;
#pragma unroll
      for (int j = 0; j < 8; ++j) {
        const int m = btile + (hi << 3) + j;
        __builtin_prefetch(&states_out[vnext + (size_t)m * NN + n], 0, 0);
      }
    }

    __syncthreads();  // one barrier/step; ping-pong covers the WAR hazard

    // u = leak_i*u + s @ (W2hi + W2lo)^T, K=512 in 16 chunks of 32
#pragma unroll
    for (int j = 0; j < 8; ++j) u[j] *= li;

    const __bf16* arow = s_lds[t & 1] + lm * NN + (hi << 3);
#pragma unroll 4
    for (int k = 0; k < NN; k += 32) {
      v8bf a0 = *(const v8bf*)(arow + k);        // ds_load_b128
      v8bf a1 = *(const v8bf*)(arow + k + 16);
      v16bf A = cat8(a0, a1);
      v8bf bh0 = *(const v8bf*)(brow_hi + k);    // global_load_b128 (L2 hit)
      v8bf bh1 = *(const v8bf*)(brow_hi + k + 16);
      v8bf bl0 = *(const v8bf*)(brow_lo + k);
      v8bf bl1 = *(const v8bf*)(brow_lo + k + 16);
      u = wmma_bf16_k32(A, cat8(bh0, bh1), u);
      u = wmma_bf16_k32(A, cat8(bl0, bl1), u);
    }

#pragma unroll
    for (int j = 0; j < 8; ++j)
      decoded_out[((size_t)t * BATCH + btile + (hi << 3) + j) * NN + n] = u[j];
  }
}

extern "C" void kernel_launch(void* const* d_in, const int* in_sizes, int n_in,
                              void* d_out, int out_size, void* d_ws,
                              size_t ws_size, hipStream_t stream) {
  (void)in_sizes; (void)n_in; (void)out_size; (void)ws_size;

  const float* X         = (const float*)d_in[0];  // [64,1000,512]
  const float* state_snn = (const float*)d_in[1];  // [2,64,512]
  const float* state_LI  = (const float*)d_in[2];  // [64,512]
  const float* W1        = (const float*)d_in[3];  // [512,512]
  const float* W2        = (const float*)d_in[4];  // [512,512]
  const float* leak_v    = (const float*)d_in[5];  // [512]
  const float* leak_i    = (const float*)d_in[6];  // [512]
  const float* thresh    = (const float*)d_in[7];  // [512]

  float* spikes_out  = (float*)d_out;                                  // [T,B,N]
  float* states_out  = spikes_out + (size_t)T_STEPS * BATCH * NN;      // [T,2,B,N]
  float* decoded_out = states_out + (size_t)T_STEPS * 2 * BATCH * NN;  // [T,B,N]

  // workspace: split-bf16 W2 (2 x 512KB = 1MB)
  __bf16* w2hi = (__bf16*)d_ws;
  __bf16* w2lo = w2hi + (size_t)NN * NN;

  w2_split_kernel<<<(NN * NN) / 256, 256, 0, stream>>>(W2, w2hi, w2lo);

  // 64000x512 @ 512x512 f32 GEMM: 4000x32 wave-tiles, 8 waves/block
  ff_gemm_kernel<<<16000, 256, 0, stream>>>(X, W1, states_out);

  // sequential scan: 4 persistent workgroups of 32 waves
  snn_scan_kernel<<<4, 1024, 0, stream>>>(state_snn, state_LI, w2hi, w2lo,
                                          leak_v, leak_i, thresh, spikes_out,
                                          states_out, decoded_out);
}
